// MultiHeadAttention_59588376264826
// MI455X (gfx1250) — matrix operand order repaired and checked by reading, compiled, not run on hardware
//
#include <hip/hip_runtime.h>

// Fixed problem dims from the reference
constexpr int B_  = 2;
constexpr int T_  = 2048;
constexpr int C_  = 1024;
constexpr int H_  = 16;
constexpr int S_  = 64;
constexpr int BT_ = B_ * T_;
constexpr size_t WOFF_ = (size_t)H_ * S_ * C_;  // one head-stacked weight matrix

typedef __attribute__((ext_vector_type(16))) __bf16 v16bf;
typedef __attribute__((ext_vector_type(8)))  __bf16 v8bf;
typedef __attribute__((ext_vector_type(8)))  float  v8f;

#ifndef __has_builtin
#define __has_builtin(x) 0
#endif
#if __has_builtin(__builtin_amdgcn_tensor_load_to_lds) && \
    __has_builtin(__builtin_amdgcn_s_wait_tensorcnt)
#define HAVE_TDM 1
#else
#define HAVE_TDM 0
#endif

__device__ __forceinline__ v8f wmma_bf16(v16bf a, v16bf b, v8f c) {
  // D = A(16x32 bf16) * B(32x16 bf16) + C(16x16 f32)
  return __builtin_amdgcn_wmma_f32_16x16x32_bf16(
      /*neg_a=*/false, a, /*neg_b=*/false, b,
      /*c_mod=*/(short)0, c, /*reuse_a=*/false, /*reuse_b=*/false);
}

// A fragment (16x32 bf16). Memory: row-major, K contiguous, row stride = ld.
// Contiguous k order used here (not AMD's table order): lanes 0-15 hold M=lane, k=0..15; lanes 16-31 hold
// M=lane-16, k=16..31 (matches load_bT_frag).
__device__ __forceinline__ v16bf load_a_frag(const __bf16* base, int ld, int lane) {
  const __bf16* p = base + (size_t)(lane & 15) * ld + ((lane >> 4) << 4);
  return *(const v16bf*)p;
}

// B fragment (32x16) where B[k][n] = W[n][k], W row-major with k contiguous.
// Contiguous k order used here: lane = column n (mod 16); lanes 0-15 hold k=0..15, lanes 16-31
// hold k=16..31, sequential within the 16 half-slots. One 32B contiguous load.
__device__ __forceinline__ v16bf load_bT_frag(const __bf16* w, int ld, int lane) {
  const __bf16* p = w + (size_t)(lane & 15) * ld + ((lane >> 4) << 4);
  return *(const v16bf*)p;
}

#if HAVE_TDM
typedef unsigned int u32_t;
typedef __attribute__((ext_vector_type(4))) u32_t v4u;
typedef __attribute__((ext_vector_type(8))) int   v8i;
typedef __attribute__((ext_vector_type(4))) int   v4i;

// TDM: DMA a fully-in-bounds 2D tile (tile_k x tile_rows, row stride
// row_stride elems, 2-byte elements) from global to LDS (row-major, packed).
__device__ __forceinline__ void tdm_load_2d_to_lds(const void* gaddr,
                                                   u32_t lds_byte_off,
                                                   u32_t tile_k, u32_t tile_rows,
                                                   u32_t row_stride) {
  unsigned long long ga = (unsigned long long)(uintptr_t)gaddr;
  v4u g0;
  g0[0] = 1u;                                   // count=1, user descriptor
  g0[1] = lds_byte_off;                         // lds_addr [63:32]
  g0[2] = (u32_t)(ga & 0xffffffffu);            // global_addr lo
  g0[3] = (u32_t)((ga >> 32) & 0x01ffffffu) | (2u << 30);  // addr hi | type=2
  v8i g1;
  g1[0] = (int)(1u << 16);                      // workgroup_mask=0, data_size=1 (2B)
  g1[1] = (int)((tile_k & 0xffffu) << 16);      // tensor_dim0[15:0] = tile_k
  g1[2] = (int)((tile_k >> 16) | ((tile_rows & 0xffffu) << 16));  // d0 hi | tensor_dim1 lo
  g1[3] = (int)((tile_rows >> 16) | ((tile_k & 0xffffu) << 16));  // d1 hi | tile_dim0
  g1[4] = (int)(tile_rows & 0xffffu);           // tile_dim1 (tile_dim2=0)
  g1[5] = (int)row_stride;                      // tensor_dim0_stride lo32
  g1[6] = 0;                                    // d0_stride hi16 | d1_stride lo16
  g1[7] = 0;
  v4i z4 = {0, 0, 0, 0};                        // groups 2/3 unused (2D tile)
  v8i z8 = {0, 0, 0, 0, 0, 0, 0, 0};
  __builtin_amdgcn_tensor_load_to_lds(g0, g1, z4, z4, z8, 0);  // 6-arg form
}
#endif

// ---------------------------------------------------------------- prep kernels
__global__ void cast_f32_bf16_kernel(const float* __restrict__ in,
                                     __bf16* __restrict__ out, int n) {
  int i = blockIdx.x * blockDim.x + threadIdx.x;
  if (i < n) out[i] = (__bf16)in[i];
}

// W[h,c,s] (f32) -> Wt[h,s,c] (bf16) so GEMM B-fragments are k(=c)-contiguous.
__global__ void transpose_w_kernel(const float* __restrict__ w,
                                   __bf16* __restrict__ wt) {
  int i = blockIdx.x * blockDim.x + threadIdx.x;  // over H*S*C
  if (i >= H_ * S_ * C_) return;
  int c = i % C_;
  int r = i / C_;
  int s = r % S_;
  int h = r / S_;
  wt[i] = (__bf16)w[((size_t)h * C_ + c) * S_ + s];
}

// ------------------------------------------------------- fused QKV projection
// One A-fragment of x feeds Q, K and V WMMAs (12 accumulators / wave).
// wt_all holds the three transposed weights contiguously ([3][H,S,C]) so all
// 12 B-fragment loads share ONE address register: K/V live at +2MB/+4MB which
// fits the signed 24-bit IOFFSET immediate of global_load (+-8MB).
// Q,K stored [B,H,T,S]; V stored transposed [B,H,S,T] so attention's PV
// B-fragments become contiguous 32B loads.
__global__ void __launch_bounds__(256, 1)
qkv_fused_kernel(const __bf16* __restrict__ xb,      // [BT, C]
                 const __bf16* __restrict__ wt_all,  // [3][H, S, C] (Q,K,V)
                 __bf16* __restrict__ Qb,            // [B,H,T,S]
                 __bf16* __restrict__ Kb,            // [B,H,T,S]
                 __bf16* __restrict__ Vt) {          // [B,H,S,T]
  int lane = threadIdx.x & 31;
  int wave = threadIdx.x >> 5;
  int hi = lane >> 4, ln = lane & 15;
  int h  = blockIdx.y;
  int m0 = blockIdx.x * 128 + wave * 16;

  const __bf16* wq = wt_all + (size_t)h * S_ * C_;  // single base; K=+WOFF_, V=+2*WOFF_

  v8f aq[4] = {{}, {}, {}, {}};
  v8f ak[4] = {{}, {}, {}, {}};
  v8f av[4] = {{}, {}, {}, {}};
  for (int k0 = 0; k0 < C_; k0 += 32) {
    v16bf a = load_a_frag(xb + (size_t)m0 * C_ + k0, C_, lane);
    __builtin_prefetch(xb + (size_t)m0 * C_ + k0 + 64, 0, 1);  // global_prefetch_b8
#pragma unroll
    for (int j = 0; j < 4; ++j) {
      const __bf16* wj = wq + (size_t)(j * 16) * C_ + k0;
      aq[j] = wmma_bf16(a, load_bT_frag(wj,             C_, lane), aq[j]);
      ak[j] = wmma_bf16(a, load_bT_frag(wj + WOFF_,     C_, lane), ak[j]);
      av[j] = wmma_bf16(a, load_bT_frag(wj + 2 * WOFF_, C_, lane), av[j]);
    }
  }

  int bb = m0 >> 11;         // m0 / T_ (tile never straddles batch boundary)
  int t0 = m0 & (T_ - 1);
#pragma unroll
  for (int r = 0; r < 8; ++r) {
    int t = t0 + r + 8 * hi;
    size_t base = (((size_t)bb * H_ + h) * T_ + t) * S_;
#pragma unroll
    for (int j = 0; j < 4; ++j) {
      Qb[base + j * 16 + ln] = (__bf16)aq[j][r];
      Kb[base + j * 16 + ln] = (__bf16)ak[j][r];
    }
  }
  size_t vbase = ((size_t)(bb * H_ + h)) * S_ * T_;
#pragma unroll
  for (int j = 0; j < 4; ++j) {
    v8bf pk;
#pragma unroll
    for (int r = 0; r < 8; ++r) pk[r] = (__bf16)av[j][r];
    *(v8bf*)(Vt + vbase + (size_t)(j * 16 + ln) * T_ + (t0 + 8 * hi)) = pk;
  }
}

// ---------------------------------------------------------- flash attention
// Each wave: 16 query rows, streams 32-key tiles with online softmax.
__global__ void __launch_bounds__(256, 1)
attn_kernel(const __bf16* __restrict__ Qb,
            const __bf16* __restrict__ Kb,
            const __bf16* __restrict__ Vt,   // [B,H,S,T]
            __bf16* __restrict__ Yb) {       // [BT, C] head-concat
  __shared__ __align__(32) __bf16 ldsP[8][16][32];  // per-wave P staging
  int lane = threadIdx.x & 31;
  int wave = threadIdx.x >> 5;
  int hi = lane >> 4, ln = lane & 15;
  int bb = blockIdx.y >> 4;
  int h  = blockIdx.y & 15;
  int q0 = blockIdx.x * 128 + wave * 16;

  const __bf16* Qh = Qb + ((size_t)(bb * H_ + h)) * T_ * S_;
  const __bf16* Kh = Kb + ((size_t)(bb * H_ + h)) * T_ * S_;
  const __bf16* Vh = Vt + ((size_t)(bb * H_ + h)) * S_ * T_;

  // Q fragments are loop-invariant: 16 rows x full S=64 (two 16x32 A-frags)
  v16bf aq0 = load_a_frag(Qh + (size_t)q0 * S_,      S_, lane);
  v16bf aq1 = load_a_frag(Qh + (size_t)q0 * S_ + 32, S_, lane);

  v8f acc0 = {}, acc1 = {}, acc2 = {}, acc3 = {};
  float mrow[8], lrow[8];
#pragma unroll
  for (int r = 0; r < 8; ++r) { mrow[r] = -1e30f; lrow[r] = 0.f; }

  int kend = q0 + 16;  // causal: keys < q0+16
  for (int kb = 0; kb < kend; kb += 32) {
    __builtin_prefetch(Kh + (size_t)(kb + 32) * S_, 0, 1);
    // scores: Q(16x64) @ K^T(64x32) as two 16x16 fragments
    v16bf bk00 = load_bT_frag(Kh + (size_t)kb * S_,             S_, lane);
    v16bf bk01 = load_bT_frag(Kh + (size_t)kb * S_ + 32,        S_, lane);
    v16bf bk10 = load_bT_frag(Kh + (size_t)(kb + 16) * S_,      S_, lane);
    v16bf bk11 = load_bT_frag(Kh + (size_t)(kb + 16) * S_ + 32, S_, lane);
    v8f s0 = {}, s1 = {};
    s0 = wmma_bf16(aq0, bk00, s0);
    s0 = wmma_bf16(aq1, bk01, s0);
    s1 = wmma_bf16(aq0, bk10, s1);
    s1 = wmma_bf16(aq1, bk11, s1);

    // online softmax; D-fragment: slot r + 8*hi = query row, lane&15 = key col
#pragma unroll
    for (int r = 0; r < 8; ++r) {
      int q = q0 + r + 8 * hi;
      int kg = kb + ln;
      float t0 = (kg      <= q) ? s0[r] * 0.125f : -1e30f;  // 1/sqrt(64)
      float t1 = (kg + 16 <= q) ? s1[r] * 0.125f : -1e30f;
      float mx = fmaxf(t0, t1);
#pragma unroll
      for (int off = 1; off < 16; off <<= 1)
        mx = fmaxf(mx, __shfl_xor(mx, off, 32));  // reduce within 16-lane half
      float mnew  = fmaxf(mrow[r], mx);
      float alpha = __expf(mrow[r] - mnew);
      mrow[r] = mnew;
      float p0 = __expf(t0 - mnew);
      float p1 = __expf(t1 - mnew);
      float rs = p0 + p1;
#pragma unroll
      for (int off = 1; off < 16; off <<= 1)
        rs += __shfl_xor(rs, off, 32);
      lrow[r] = lrow[r] * alpha + rs;
      acc0[r] *= alpha; acc1[r] *= alpha; acc2[r] *= alpha; acc3[r] *= alpha;
      // stage P (D-layout -> row-major) for A-fragment reload
      ldsP[wave][r + 8 * hi][ln]      = (__bf16)p0;
      ldsP[wave][r + 8 * hi][ln + 16] = (__bf16)p1;
    }

    // PV: P(16x32) @ V(32x64); V is transposed so B-frags are contiguous 32B.
    // Same-wave LDS RAW is in-order per the DS spec (no barrier needed).
    v16bf pa = load_a_frag(&ldsP[wave][0][0], 32, lane);
    acc0 = wmma_bf16(pa, load_bT_frag(Vh + (size_t)0  * T_ + kb, T_, lane), acc0);
    acc1 = wmma_bf16(pa, load_bT_frag(Vh + (size_t)16 * T_ + kb, T_, lane), acc1);
    acc2 = wmma_bf16(pa, load_bT_frag(Vh + (size_t)32 * T_ + kb, T_, lane), acc2);
    acc3 = wmma_bf16(pa, load_bT_frag(Vh + (size_t)48 * T_ + kb, T_, lane), acc3);
  }

#pragma unroll
  for (int r = 0; r < 8; ++r) {
    float inv = 1.f / lrow[r];
    int t = q0 + r + 8 * hi;
    size_t row = ((size_t)bb * T_ + t) * C_ + (size_t)h * S_;
    Yb[row + 0  + ln] = (__bf16)(acc0[r] * inv);
    Yb[row + 16 + ln] = (__bf16)(acc1[r] * inv);
    Yb[row + 32 + ln] = (__bf16)(acc2[r] * inv);
    Yb[row + 48 + ln] = (__bf16)(acc3[r] * inv);
  }
}

// ------------------------------------------------------------ output proj
// out[m,n] = sum_c y[m,c] * Wp[n,c] + bp[n]   (y @ Wp^T + bp)
// All 8 waves share the same 64x32 Wp tile per k-step -> stage it once via the
// Tensor Data Mover (wave 0 issues, s_wait_tensorcnt, barrier publishes).
__global__ void __launch_bounds__(256, 1)
out_proj_kernel(const __bf16* __restrict__ yb,   // [BT, C]
                const __bf16* __restrict__ wp,   // [C, C] (n,c)
                const float* __restrict__ bp,
                float* __restrict__ out) {       // [BT, C]
  int lane = threadIdx.x & 31;
  int wave = threadIdx.x >> 5;
  int hi = lane >> 4, ln = lane & 15;
  int m0 = blockIdx.x * 128 + wave * 16;
  int n0 = blockIdx.y * 64;

  v8f acc[4] = {{}, {}, {}, {}};
#if HAVE_TDM
  __shared__ __align__(16) __bf16 ldsB[64][32];  // Wp[n0..n0+63][k0..k0+31]
  u32_t lds_off = (u32_t)(uintptr_t)(__attribute__((address_space(3))) void*)&ldsB[0][0];
  for (int k0 = 0; k0 < C_; k0 += 32) {
    if (wave == 0) {
      tdm_load_2d_to_lds(wp + (size_t)n0 * C_ + k0, lds_off,
                         /*tile_k=*/32, /*tile_rows=*/64, /*row_stride=*/C_);
      __builtin_amdgcn_s_wait_tensorcnt(0);
    }
    __syncthreads();
    v16bf a = load_a_frag(yb + (size_t)m0 * C_ + k0, C_, lane);
    __builtin_prefetch(yb + (size_t)m0 * C_ + k0 + 64, 0, 1);
#pragma unroll
    for (int j = 0; j < 4; ++j)
      acc[j] = wmma_bf16(a, load_bT_frag(&ldsB[j * 16][0], 32, lane), acc[j]);
    __syncthreads();
  }
#else
  for (int k0 = 0; k0 < C_; k0 += 32) {
    v16bf a = load_a_frag(yb + (size_t)m0 * C_ + k0, C_, lane);
#pragma unroll
    for (int j = 0; j < 4; ++j)
      acc[j] = wmma_bf16(a, load_bT_frag(wp + (size_t)(n0 + j * 16) * C_ + k0, C_, lane), acc[j]);
  }
#endif

#pragma unroll
  for (int j = 0; j < 4; ++j) {
    float bias = bp[n0 + j * 16 + ln];
#pragma unroll
    for (int r = 0; r < 8; ++r) {
      size_t m = (size_t)(m0 + r + 8 * hi);
      out[m * C_ + n0 + j * 16 + ln] = acc[j][r] + bias;
    }
  }
}

extern "C" void kernel_launch(void* const* d_in, const int* in_sizes, int n_in,
                              void* d_out, int out_size, void* d_ws, size_t ws_size,
                              hipStream_t stream) {
  (void)in_sizes; (void)n_in; (void)out_size; (void)ws_size;
  const float* x  = (const float*)d_in[0];
  const float* Wq = (const float*)d_in[1];
  const float* Wk = (const float*)d_in[2];
  const float* Wv = (const float*)d_in[3];
  const float* Wp = (const float*)d_in[4];
  const float* bp = (const float*)d_in[5];
  float* out = (float*)d_out;

  // workspace carve-up (bf16 elements), total ~48 MB
  // NOTE: wqt/wkt/wvt must stay contiguous (qkv_fused uses immediate offsets).
  __bf16* ws = (__bf16*)d_ws;
  size_t o = 0;
  __bf16* xb  = ws + o; o += (size_t)BT_ * C_;
  __bf16* wqt = ws + o; o += WOFF_;
  __bf16* wkt = ws + o; o += WOFF_;
  __bf16* wvt = ws + o; o += WOFF_;
  __bf16* wpb = ws + o; o += (size_t)C_ * C_;
  __bf16* Qb  = ws + o; o += (size_t)B_ * H_ * T_ * S_;
  __bf16* Kb  = ws + o; o += (size_t)B_ * H_ * T_ * S_;
  __bf16* Vt  = ws + o; o += (size_t)B_ * H_ * S_ * T_;
  __bf16* Yb  = ws + o; o += (size_t)BT_ * C_;

  // 1) precision prep
  cast_f32_bf16_kernel<<<(BT_ * C_ + 255) / 256, 256, 0, stream>>>(x, xb, BT_ * C_);
  transpose_w_kernel<<<(H_ * S_ * C_ + 255) / 256, 256, 0, stream>>>(Wq, wqt);
  transpose_w_kernel<<<(H_ * S_ * C_ + 255) / 256, 256, 0, stream>>>(Wk, wkt);
  transpose_w_kernel<<<(H_ * S_ * C_ + 255) / 256, 256, 0, stream>>>(Wv, wvt);
  cast_f32_bf16_kernel<<<(C_ * C_ + 255) / 256, 256, 0, stream>>>(Wp, wpb, C_ * C_);

  // 2) fused QKV projection (WMMA), V written transposed
  dim3 gq(BT_ / 128, H_);
  qkv_fused_kernel<<<gq, 256, 0, stream>>>(xb, wqt, Qb, Kb, Vt);

  // 3) causal flash attention (WMMA)
  dim3 ga(T_ / 128, B_ * H_);
  attn_kernel<<<ga, 256, 0, stream>>>(Qb, Kb, Vt, Yb);

  // 4) output projection + bias (WMMA, TDM-staged weight tile)
  dim3 go(BT_ / 128, C_ / 64);
  out_proj_kernel<<<go, 256, 0, stream>>>(Yb, wpb, bp, out);
}
